// Wconv_53884659695887
// MI455X (gfx1250) — compile-verified, matrix operand order checked by reading, not run on hardware
//
#include <hip/hip_runtime.h>
#include <stdint.h>

typedef __attribute__((ext_vector_type(16))) _Float16     v16h;
typedef __attribute__((ext_vector_type(2)))  _Float16     v2h;
typedef __attribute__((ext_vector_type(8)))  float        v8f;
typedef __attribute__((ext_vector_type(4)))  unsigned int v4u;
typedef __attribute__((ext_vector_type(8)))  int          v8i;
typedef __attribute__((ext_vector_type(4)))  int          v4i;

#define F_DIM   512
#define HDIM    256
#define N_SAMP  8
#define OUTC    64
#define INC     64
#define KSZ     3
#define KTOT    (INC * KSZ * KSZ)      // 576
#define BATCH   4
#define IMGH    128
#define IMGW    128
#define HWPIX   (IMGH * IMGW)          // 16384
#define M_TOTAL (N_SAMP * OUTC)        // 512
#define PIXTOT  (BATCH * HWPIX)        // 65536

#define BLK_M   32
#define BLK_P   128                    // one full image row per block
#define KCH     32
#define NKC     (KTOT / KCH)           // 18 (= 9 kernel positions x 2 ic-halves)
#define BSTR    48                     // padded f16 stride of B tile rows (px-major)

#if defined(__has_builtin)
# if __has_builtin(__builtin_amdgcn_tensor_load_to_lds) && __has_builtin(__builtin_amdgcn_s_wait_tensorcnt)
#  define USE_TDM 1
# endif
#endif
#ifndef USE_TDM
# define USE_TDM 0
#endif

// K-order permutation shared by A (weights) and B (im2col):
//   kperm = (ky*3+kx)*64 + ic      so chunk kc covers fixed (ky,kx),
//   ic in [ (kc&1)*32, (kc&1)*32+32 ),  kpos = kc>>1.

// ---------------------------------------------------------------------------
// Kernel 1: h = relu(features @ W1^T + b1)   [8, 256]
// ---------------------------------------------------------------------------
__global__ void mlp_h_kernel(const float* __restrict__ feat,
                             const float* __restrict__ W1,
                             const float* __restrict__ b1,
                             float* __restrict__ h) {
  const int n = blockIdx.x;
  const int j = threadIdx.x;
  const float* fr = feat + n * F_DIM;
  const float* wr = W1 + (size_t)j * F_DIM;
  float acc = b1[j];
  for (int f = 0; f < F_DIM; ++f) acc = fmaf(fr[f], wr[f], acc);
  h[n * HDIM + j] = acc > 0.f ? acc : 0.f;
}

// ---------------------------------------------------------------------------
// Kernel 2: per-filter (n,oc): w = h@W2^T + b2 ; LayerNorm over 576 ; *gamma
// + beta ; /inC ; quantize to f16 into K-PERMUTED layout Wq[512][576]
// ---------------------------------------------------------------------------
__global__ void weight_kernel(const float* __restrict__ h,
                              const float* __restrict__ W2,
                              const float* __restrict__ b2,
                              const float* __restrict__ gamma,
                              const float* __restrict__ beta,
                              _Float16* __restrict__ wq) {
  __shared__ float sh[HDIM];
  __shared__ float rs[256];
  __shared__ float rq[256];
  __shared__ float s_mean, s_rstd;

  const int m   = blockIdx.x;          // 0..511 filter index
  const int n   = m / OUTC;
  const int oc  = m % OUTC;
  const int tid = threadIdx.x;

  sh[tid] = h[n * HDIM + tid];
  __syncthreads();

  float lv[3];
  float lsum = 0.f, lsq = 0.f;
  int cnt = 0;
  for (int k = tid; k < KTOT; k += 256) {
    const size_t row = (size_t)(oc * KTOT + k);
    const float* wrow = W2 + row * HDIM;
    float acc = b2[row];
    for (int c = 0; c < HDIM; ++c) acc = fmaf(sh[c], wrow[c], acc);
    lv[cnt++] = acc;
    lsum += acc;
    lsq  = fmaf(acc, acc, lsq);
  }
  rs[tid] = lsum;
  rq[tid] = lsq;
  __syncthreads();
  for (int s = 128; s > 0; s >>= 1) {
    if (tid < s) { rs[tid] += rs[tid + s]; rq[tid] += rq[tid + s]; }
    __syncthreads();
  }
  if (tid == 0) {
    const float mean = rs[0] * (1.f / KTOT);
    const float var  = rq[0] * (1.f / KTOT) - mean * mean;   // biased, like jnp.var
    s_mean = mean;
    s_rstd = rsqrtf(var + 1e-5f);
  }
  __syncthreads();
  const float mean = s_mean, rstd = s_rstd;
  cnt = 0;
  for (int k = tid; k < KTOT; k += 256) {
    float wn = (lv[cnt++] - mean) * rstd * gamma[k] + beta[k];
    const int ic = k / 9;              // original flat k = ic*9 + kpos
    const int kp = k - ic * 9;
    wq[(size_t)m * KTOT + kp * INC + ic] = (_Float16)(wn * (1.f / INC));
  }
}

// ---------------------------------------------------------------------------
// Kernel 3: implicit-GEMM conv via v_wmma_f32_16x16x32_f16
//   M = 512 filters, Kperm = 576, Npix = 65536 (B*H*W)
//   Block tile: 32 M x 128 px (one image row), 8 waves = 2(M) x 4(px pairs),
//   each wave: 1 A-frag x 2 accumulators per K-chunk of 32.
//   A tile staged by TDM; B tile im2col (fixed kernel pos per chunk,
//   divide-free, batched loads) double-buffered.
// ---------------------------------------------------------------------------
__global__ void __launch_bounds__(256)
wconv_kernel(const float* __restrict__ in,
             const _Float16* __restrict__ wq,
             float* __restrict__ out) {
  __shared__ __align__(128) _Float16 Alds[BLK_M * KTOT];        // 36864 B
  __shared__ __align__(128) _Float16 Blds[2][BLK_P * BSTR];     // 2 x 12288 B

  const int tid   = threadIdx.x;
  const int lane  = tid & 31;
  const int wave  = tid >> 5;
  const int wm    = wave >> 2;           // 0..1  (M sub-tile)
  const int wp    = wave & 3;            // 0..3  (pixel sub-tile pair)
  const int mtile = blockIdx.y;

  const unsigned P0   = blockIdx.x * BLK_P;   // first pixel of this row-tile
  const int bimg      = P0 >> 14;             // / HWPIX
  const int rowoff    = P0 & (HWPIX - 1);     // y*W
  const int y         = rowoff >> 7;          // / IMGW

  // ---- Stage A tile: 32 filters x 576 Kperm of f16 weights ----------------
#if USE_TDM
  if (tid == 0) {
    const unsigned long long ga =
        (unsigned long long)(uintptr_t)(wq + (size_t)mtile * BLK_M * KTOT);
    const unsigned ldsa = (unsigned)(uintptr_t)&Alds[0];
    // D# group 0: count=1 | lds_addr | global_addr | type=2
    v4u g0 = { 1u, ldsa, (unsigned)ga,
               ((unsigned)((ga >> 32) & 0x01FFFFFFu)) | (2u << 30) };
    // D# group 1: data_size=2B; tensor_dim0=576; tensor_dim1=32;
    //             tile_dim0=576; tile_dim1=32; dim0_stride=576
    v8i g1 = { (int)0x00010000,
               (int)((unsigned)KTOT  << 16),
               (int)((unsigned)BLK_M << 16),
               (int)((unsigned)KTOT  << 16),
               BLK_M,
               KTOT,
               0, 0 };
    v4i gz4 = { 0, 0, 0, 0 };
    v8i gz8 = { 0, 0, 0, 0, 0, 0, 0, 0 };
    __builtin_amdgcn_tensor_load_to_lds(g0, g1, gz4, gz4, gz8, 0);
  }
#else
  {
    const unsigned* src = (const unsigned*)(wq + (size_t)mtile * BLK_M * KTOT);
    unsigned* dst = (unsigned*)&Alds[0];
    for (int i = tid; i < (BLK_M * KTOT) / 2; i += 256) dst[i] = src[i];
  }
#endif

  // ---- im2col B-tile stager -----------------------------------------------
  // Chunk kc: kernel position kpos = kc>>1 (fixed ky,kx), ic-half = kc&1.
  // Thread (px = tid&127, kh = tid>>7) fills kl = kh*16 .. kh*16+15 =
  // channels icb..icb+15 at (iy, ix). 16 independent loads, stride HWPIX.
  auto stageB = [&](int kc, int buf) {
    const int px   = tid & (BLK_P - 1);
    const int kh   = tid >> 7;                   // 0 or 1
    const int kpos = kc >> 1;                    // 0..8 (uniform)
    const int ky   = kpos / 3;                   // scalar divide, per chunk
    const int kx   = kpos - ky * 3;
    const int icb  = (kc & 1) * 32 + kh * 16;
    const int iy   = y + ky - 1;                 // block-uniform
    const int ix   = px + kx - 1;                // per-lane (2/128 OOB max)

    float v[16];
#pragma unroll
    for (int i = 0; i < 16; ++i) v[i] = 0.f;
    if (((unsigned)iy < IMGH) & ((unsigned)ix < IMGW)) {
      const float* src =
          in + (((size_t)bimg * INC + icb) * IMGH + iy) * IMGW + ix;
#pragma unroll
      for (int i = 0; i < 16; ++i) v[i] = src[(size_t)i * HWPIX];
    }
    _Float16* dst = &Blds[buf][px * BSTR + kh * 16];
#pragma unroll
    for (int i = 0; i < 16; i += 2) {
      v2h p;
      p.x = (_Float16)v[i];
      p.y = (_Float16)v[i + 1];
      *(v2h*)(dst + i) = p;
    }
  };

  stageB(0, 0);
#if USE_TDM
  if (tid == 0) __builtin_amdgcn_s_wait_tensorcnt(0);
#endif
  __syncthreads();

  // ---- main K loop: double-buffered B, 2 WMMA per wave per chunk ----------
  v8f acc0 = {};
  v8f acc1 = {};
  const int arow  = (wm * 16 + (lane & 15)) * KTOT + ((lane >> 4) << 4);
  const int bcol0 = (wp * 32 + (lane & 15)) * BSTR + ((lane >> 4) << 4);
  const int bcol1 = bcol0 + 16 * BSTR;

  for (int kc = 0; kc < NKC; ++kc) {
    if (kc + 1 < NKC) stageB(kc + 1, (kc + 1) & 1);
    const v16h a  = *(const v16h*)&Alds[arow + kc * KCH];
    const v16h b0 = *(const v16h*)&Blds[kc & 1][bcol0];
    const v16h b1 = *(const v16h*)&Blds[kc & 1][bcol1];
    acc0 = __builtin_amdgcn_wmma_f32_16x16x32_f16(
        false, a, false, b0, (short)0, acc0, false, false);
    acc1 = __builtin_amdgcn_wmma_f32_16x16x32_f16(
        false, a, false, b1, (short)0, acc1, false, false);
    __syncthreads();
  }

  // ---- store: D layout VGPR r, lane L -> M = r + 8*(L>>4), N = L&15 -------
  const int mb  = mtile * BLK_M + wm * 16 + ((lane >> 4) << 3);
  const int xb0 = wp * 32 + (lane & 15);
#pragma unroll
  for (int r = 0; r < 8; ++r) {
    const int m    = mb + r;
    const int nidx = m >> 6;        // sample n
    const int oc   = m & 63;
    const size_t base =
        (((size_t)(nidx * BATCH + bimg) * OUTC + oc) * HWPIX) + rowoff;
    out[base + xb0]      = acc0[r];
    out[base + xb0 + 16] = acc1[r];
  }
}

// ---------------------------------------------------------------------------
extern "C" void kernel_launch(void* const* d_in, const int* in_sizes, int n_in,
                              void* d_out, int out_size, void* d_ws, size_t ws_size,
                              hipStream_t stream) {
  (void)in_sizes; (void)n_in; (void)out_size; (void)ws_size;
  const float* input = (const float*)d_in[0];
  const float* feat  = (const float*)d_in[1];
  const float* W1    = (const float*)d_in[2];
  const float* b1    = (const float*)d_in[3];
  const float* W2    = (const float*)d_in[4];
  const float* b2    = (const float*)d_in[5];
  const float* gamma = (const float*)d_in[6];
  const float* beta  = (const float*)d_in[7];
  float* out = (float*)d_out;

  float*    h  = (float*)d_ws;                          // 8*256 fp32  (8 KB)
  _Float16* wq = (_Float16*)((char*)d_ws + 8192);       // 512*576 f16 (576 KB)

  mlp_h_kernel<<<dim3(N_SAMP), dim3(HDIM), 0, stream>>>(feat, W1, b1, h);
  weight_kernel<<<dim3(M_TOTAL), dim3(256), 0, stream>>>(h, W2, b2, gamma, beta, wq);
  wconv_kernel<<<dim3(PIXTOT / BLK_P, M_TOTAL / BLK_M), dim3(256), 0, stream>>>(
      input, wq, out);
}
